// TemporalInteraction_23158463660309
// MI455X (gfx1250) — compile-verified, matrix operand order checked by reading, not run on hardware
//
#include <hip/hip_runtime.h>
#include <hip/hip_bf16.h>
#include <stdint.h>

// ---------------------------------------------------------------------------
// TemporalInteraction on MI455X (gfx1250): all MLPs run as bf16 WMMA GEMMs
// (v_wmma_f32_16x16x32_bf16, f32 accumulate), edge gathers staged through LDS.
// Round 2: A fragments held in registers across n-tiles; K loop fully
// unrolled so B loads clause together and waits are amortized per n-tile.
// ---------------------------------------------------------------------------

typedef __attribute__((ext_vector_type(16))) __bf16 v16bf;
typedef __attribute__((ext_vector_type(8)))  float  v8f;

union ABu { v16bf v; uint32_t u[8]; };

static constexpr int Bb = 4, Nn = 16384, Mm = 4096;
static constexpr int BN = Bb * Nn;   // 65536 nodes
static constexpr int BM = Bb * Mm;   // 16384 queries

// ---------------------------------------------------------------------------
// Weight packing: fp32 W[dinTotal, dout] (row-major, x @ W) -> bf16 fragments.
// Fragment = (kchunk, ntile) of 32(K) x 16(N); 512 bf16 elems, lane-major
// (lane*16 + e), element e mapped per the 16-bit A-matrix 16x32 VGPR layout
// (CDNA5 ISA 7.12.2) applied to W^T. Zero rows are inserted where
// the concat halves were padded: logical k<dA -> row k; k in [dApad,dApad+dB)
// -> row dA+(k-dApad); otherwise 0.
// ---------------------------------------------------------------------------
__global__ void k_pack_w(const float* __restrict__ W, __bf16* __restrict__ dst,
                         int dA, int dApad, int dB, int dout, int KDIM)
{
    const int NT = dout / 16;
    const long total = (long)(KDIM / 32) * NT * 512;
    const long stride = (long)gridDim.x * blockDim.x;
    for (long g = (long)blockIdx.x * blockDim.x + threadIdx.x; g < total; g += stride) {
        const int  e    = (int)(g & 15);
        const int  lane = (int)((g >> 4) & 31);
        const long frag = g >> 9;
        const int  nt   = (int)(frag % NT);
        const int  kc   = (int)(frag / NT);
        const int  hi   = lane >> 4;
        const int  vg   = e >> 1, half = e & 1;
        const int  k    = ((vg < 4 ? vg * 2 : 16 + (vg - 4) * 2) + hi * 8 + half) + kc * 32;
        const int  n    = nt * 16 + (lane & 15);
        float val = 0.f;
        int row = -1;
        if (k < dA)                                   row = k;
        else if (k >= dApad && (k - dApad) < dB)      row = dA + (k - dApad);
        if (row >= 0) val = W[(long)row * dout + n];
        dst[g] = (__bf16)val;
    }
}

// ---------------------------------------------------------------------------
// Time encoding: t -> [sin(t*{1,2,4,8}), cos(t*{1,2,4,8})]
// ---------------------------------------------------------------------------
__global__ void k_enc_time(const float* __restrict__ tp, float* __restrict__ et, int R)
{
    int g = blockIdx.x * 256 + threadIdx.x;
    if (g >= R) return;
    const float t = tp[g];
    const float fr[4] = {1.f, 2.f, 4.f, 8.f};
#pragma unroll
    for (int j = 0; j < 4; ++j) {
        et[g * 8 + j]     = sinf(t * fr[j]);
        et[g * 8 + 4 + j] = cosf(t * fr[j]);
    }
}

// query time encoding, padded to 16 cols, bf16
__global__ void k_enc_query(const float* __restrict__ qp, __bf16* __restrict__ q, int R)
{
    int g = blockIdx.x * 256 + threadIdx.x;
    if (g >= R) return;
    const float t = qp[g];
    const float fr[4] = {1.f, 2.f, 4.f, 8.f};
#pragma unroll
    for (int j = 0; j < 4; ++j) {
        q[g * 16 + j]     = (__bf16)sinf(t * fr[j]);
        q[g * 16 + 4 + j] = (__bf16)cosf(t * fr[j]);
    }
#pragma unroll
    for (int j = 8; j < 16; ++j) q[g * 16 + j] = (__bf16)0.f;
}

// ---------------------------------------------------------------------------
// Concat up to 3 fp32 sources into a padded bf16 node-feature buffer [R, DP].
// Padding columns are zero-filled.
// ---------------------------------------------------------------------------
__global__ void k_concat3(__bf16* __restrict__ dst, int DP, long R,
                          const float* __restrict__ s0, int d0,
                          const float* __restrict__ s1, int d1,
                          const float* __restrict__ s2, int d2)
{
    const long total = R * DP;
    const long stride = (long)gridDim.x * blockDim.x;
    for (long g = (long)blockIdx.x * blockDim.x + threadIdx.x; g < total; g += stride) {
        const long r = g / DP;
        const int  c = (int)(g % DP);
        float v = 0.f;
        if (c < d0)                 v = s0[r * d0 + c];
        else if (c < d0 + d1)       v = s1[r * d1 + (c - d0)];
        else if (c < d0 + d1 + d2)  v = s2[r * d2 + (c - d0 - d1)];
        dst[g] = (__bf16)v;
    }
}

// ---------------------------------------------------------------------------
// Load this wave's A fragments (16-row strip, all K chunks) from the LDS tile
// into registers once; they are reused for every output n-tile.
// ---------------------------------------------------------------------------
template <int KD>
__device__ __forceinline__ void load_a_frags(const __bf16* __restrict__ Asrc,
                                             ABu (&af)[KD / 32])
{
    const int tid  = threadIdx.x;
    const int wave = tid >> 5;
    const int lane = tid & 31;
    const int hi   = lane >> 4;
    const int lcol = lane & 15;
    const uint32_t* arow =
        (const uint32_t*)(Asrc + (size_t)(wave * 16 + lcol) * KD);
#pragma unroll
    for (int kc = 0; kc < KD / 32; ++kc) {
#pragma unroll
        for (int e = 0; e < 8; ++e)
            af[kc].u[e] = arow[kc * 16 + (e < 4 ? e : e + 4) + hi * 4];
    }
}

// ---------------------------------------------------------------------------
// One 16-row-strip GEMM layer: A fragments (registers) x packed W -> Dst bf16.
// K loop fully unrolled: all B-fragment global_load_b128s issue as a clause,
// then the chained WMMAs run behind a single wait.
// ---------------------------------------------------------------------------
template <int KD, int NOUT, bool RELU>
__device__ __forceinline__ void gemm_bf16(const __bf16* __restrict__ Asrc,
                                          const __bf16* __restrict__ W,
                                          const float* __restrict__ Bias,
                                          __bf16* __restrict__ Dst)
{
    const int tid  = threadIdx.x;
    const int wave = tid >> 5;
    const int lane = tid & 31;
    const int hi   = lane >> 4;
    const int lcol = lane & 15;
    const int rowBase = wave * 16;

    ABu af[KD / 32];
    load_a_frags<KD>(Asrc, af);

#pragma unroll 1
    for (int nt = 0; nt < NOUT / 16; ++nt) {
        v8f acc = {0.f, 0.f, 0.f, 0.f, 0.f, 0.f, 0.f, 0.f};
#pragma unroll
        for (int kc = 0; kc < KD / 32; ++kc) {
            ABu b;
            const uint32_t* wp =
                (const uint32_t*)(W + ((size_t)(kc * (NOUT / 16) + nt)) * 512 + lane * 16);
#pragma unroll
            for (int e = 0; e < 8; ++e) b.u[e] = wp[e];
            acc = __builtin_amdgcn_wmma_f32_16x16x32_bf16(false, af[kc].v, false, b.v,
                                                          (short)0, acc, false, false);
        }
        const float bias = Bias[nt * 16 + lcol];
#pragma unroll
        for (int v = 0; v < 8; ++v) {
            float x = acc[v] + bias;
            if (RELU) x = fmaxf(x, 0.f);
            Dst[(size_t)(rowBase + hi * 8 + v) * NOUT + nt * 16 + lcol] = (__bf16)x;
        }
    }
}

// Final layer: same GEMM, but write f32 edge outputs scaled by validity mask.
template <int KD, int NOUT>
__device__ __forceinline__ void gemm_bf16_out(const __bf16* __restrict__ Asrc,
                                              const __bf16* __restrict__ W,
                                              const float* __restrict__ Bias,
                                              const float* __restrict__ vmask,
                                              float* __restrict__ Dst)
{
    const int tid  = threadIdx.x;
    const int wave = tid >> 5;
    const int lane = tid & 31;
    const int hi   = lane >> 4;
    const int lcol = lane & 15;
    const int rowBase = wave * 16;

    ABu af[KD / 32];
    load_a_frags<KD>(Asrc, af);

#pragma unroll 1
    for (int nt = 0; nt < NOUT / 16; ++nt) {
        v8f acc = {0.f, 0.f, 0.f, 0.f, 0.f, 0.f, 0.f, 0.f};
#pragma unroll
        for (int kc = 0; kc < KD / 32; ++kc) {
            ABu b;
            const uint32_t* wp =
                (const uint32_t*)(W + ((size_t)(kc * (NOUT / 16) + nt)) * 512 + lane * 16);
#pragma unroll
            for (int e = 0; e < 8; ++e) b.u[e] = wp[e];
            acc = __builtin_amdgcn_wmma_f32_16x16x32_bf16(false, af[kc].v, false, b.v,
                                                          (short)0, acc, false, false);
        }
        const float bias = Bias[nt * 16 + lcol];
#pragma unroll
        for (int v = 0; v < 8; ++v) {
            const int m = rowBase + hi * 8 + v;
            Dst[(size_t)m * NOUT + nt * 16 + lcol] = (acc[v] + bias) * vmask[m];
        }
    }
}

constexpr size_t smem_bytes(int KDIM, int H1, int H2, int DOUT)
{
    size_t l0 = (size_t)128 * 2 * (KDIM > H2 ? KDIM : H2);
    size_t l1 = (size_t)128 * ((2 * H1 > 4 * DOUT) ? 2 * H1 : 4 * DOUT);
    return l0 + l1 + 128 * sizeof(float);
}

// ---------------------------------------------------------------------------
// Generic edge-MLP stage. 128 edge rows per workgroup; KN edges per node
// (KN in {1,8,16} so whole nodes live in one tile -> direct reduction).
//   selfF: [selfRows, SDP] bf16 (padded); neiF: [B*NEI_PB, NDP] bf16 (padded)
//   idx/valid: [selfRows, KN] ints (idx values are within-batch)
//   out: [selfRows, DOUT] f32, masked-summed over KN.
// ---------------------------------------------------------------------------
template <int KN, int SDP, int NDP, int H1, int H2, int DOUT, int SELF_PB, int NEI_PB>
__global__ __launch_bounds__(256)
void edge_mlp(const __bf16* __restrict__ selfF, const __bf16* __restrict__ neiF,
              const int* __restrict__ idx, const int* __restrict__ valid,
              const __bf16* __restrict__ W1, const float* __restrict__ B1,
              const __bf16* __restrict__ W2, const float* __restrict__ B2,
              const __bf16* __restrict__ W3, const float* __restrict__ B3,
              float* __restrict__ out)
{
    constexpr int KDIM  = SDP + NDP;
    constexpr int ROWS  = 128;
    constexpr int NODES = ROWS / KN;
    constexpr size_t L0 = (size_t)ROWS * 2 * (KDIM > H2 ? KDIM : H2);
    constexpr size_t L1 = (size_t)ROWS * ((2 * H1 > 4 * DOUT) ? 2 * H1 : 4 * DOUT);

    extern __shared__ char smem[];
    __bf16* A    = (__bf16*)smem;              // [ROWS][KDIM]; reused as H2 out
    __bf16* Hb   = (__bf16*)(smem + L0);       // [ROWS][H1];  reused as f32 [ROWS][DOUT]
    float*  vmsk = (float*)(smem + L0 + L1);   // [ROWS]

    const int  tid      = threadIdx.x;
    const long nodeBase = (long)blockIdx.x * NODES;

    // ---- build the gathered/concatenated A tile (2 threads per edge row) ----
    for (int r2 = tid; r2 < ROWS * 2; r2 += 256) {
        const int  r    = r2 >> 1;
        const int  half = r2 & 1;
        const long node = nodeBase + r / KN;
        if (half == 0) {
            const uint32_t* src = (const uint32_t*)(selfF + node * SDP);
            uint32_t* dst = (uint32_t*)(A + (size_t)r * KDIM);
#pragma unroll
            for (int c = 0; c < SDP / 2; ++c) dst[c] = src[c];
            float v = 1.f;
            if (KN > 1 && valid) v = valid[node * KN + (r % KN)] ? 1.f : 0.f;
            vmsk[r] = v;
        } else if constexpr (NDP > 0) {
            const int  b  = (int)(node / SELF_PB);
            const int  ni = idx[node * KN + (r % KN)];
            const uint32_t* src = (const uint32_t*)(neiF + ((long)b * NEI_PB + ni) * NDP);
            uint32_t* dst = (uint32_t*)(A + (size_t)r * KDIM + SDP);
#pragma unroll
            for (int c = 0; c < NDP / 2; ++c) dst[c] = src[c];
        }
    }
    __syncthreads();

    gemm_bf16<KDIM, H1, true>(A, W1, B1, Hb);             // layer 1 + ReLU
    __syncthreads();
    gemm_bf16<H1, H2, true>(Hb, W2, B2, A);               // layer 2 + ReLU (into old A)
    __syncthreads();
    gemm_bf16_out<H2, DOUT>(A, W3, B3, vmsk, (float*)Hb); // layer 3, masked, f32
    __syncthreads();

    // ---- reduce over KN edges per node ----
    const float* Hf = (const float*)Hb;
    for (int i = tid; i < NODES * DOUT; i += 256) {
        const int nl = i / DOUT, c = i % DOUT;
        float s = 0.f;
#pragma unroll
        for (int k = 0; k < KN; ++k) s += Hf[(size_t)(nl * KN + k) * DOUT + c];
        out[(nodeBase + nl) * DOUT + c] = s;
    }
}

// ---------------------------------------------------------------------------
// Host driver
// ---------------------------------------------------------------------------
extern "C" void kernel_launch(void* const* d_in, const int* in_sizes, int n_in,
                              void* d_out, int out_size, void* d_ws, size_t ws_size,
                              hipStream_t stream)
{
    (void)in_sizes; (void)n_in; (void)out_size; (void)ws_size;

    const float* data        = (const float*)d_in[0];
    const float* space_pts   = (const float*)d_in[2];
    const float* time_pts    = (const float*)d_in[3];
    const float* query_pts   = (const float*)d_in[4];
    const int*   space_idx   = (const int*)d_in[5];
    const int*   space_valid = (const int*)d_in[6];
    const int*   time_idx    = (const int*)d_in[7];
    const int*   time_valid  = (const int*)d_in[8];
    const int*   target_idx  = (const int*)d_in[9];
    const int*   target_valid= (const int*)d_in[10];

    // params: JAX pytree flatten of dict -> sorted keys:
    //   comb0, comb1, space0, space1, target, time0, time1; each (W,b) x 3.
    auto P = [&](int i) { return (const float*)d_in[11 + i]; };
    const float *c0W1=P(0),*c0b1=P(1),*c0W2=P(2),*c0b2=P(3),*c0W3=P(4),*c0b3=P(5);
    const float *c1W1=P(6),*c1b1=P(7),*c1W2=P(8),*c1b2=P(9),*c1W3=P(10),*c1b3=P(11);
    const float *s0W1=P(12),*s0b1=P(13),*s0W2=P(14),*s0b2=P(15),*s0W3=P(16),*s0b3=P(17);
    const float *s1W1=P(18),*s1b1=P(19),*s1W2=P(20),*s1b2=P(21),*s1W3=P(22),*s1b3=P(23);
    const float *tgW1=P(24),*tgb1=P(25),*tgW2=P(26),*tgb2=P(27),*tgW3=P(28),*tgb3=P(29);
    const float *t0W1=P(30),*t0b1=P(31),*t0W2=P(32),*t0b2=P(33),*t0W3=P(34),*t0b3=P(35);
    const float *t1W1=P(36),*t1b1=P(37),*t1W2=P(38),*t1b2=P(39),*t1W3=P(40),*t1b3=P(41);

    // workspace allocator
    char* ws = (char*)d_ws;
    size_t off = 0;
    auto alloc = [&](size_t bytes) -> void* {
        void* p = ws + off;
        off = (off + bytes + 255) & ~(size_t)255;
        return p;
    };

    float*  et    = (float*)alloc((size_t)BN * 8 * 4);
    float*  sbuf  = (float*)alloc((size_t)BN * 32 * 4);
    float*  tbuf  = (float*)alloc((size_t)BN * 32 * 4);
    float*  key   = (float*)alloc((size_t)BN * 64 * 4);
    __bf16* nodeA = (__bf16*)alloc((size_t)BN * 128 * 2);
    __bf16* qenc  = (__bf16*)alloc((size_t)BM * 16 * 2);

    auto walloc = [&](int kdim, int dout) {
        return (__bf16*)alloc((size_t)kdim * dout * 2);
    };
    __bf16 *s0P1=walloc(96,64),  *s0P2=walloc(64,64),  *s0P3=walloc(64,32);
    __bf16 *t0P1=walloc(160,64), *t0P2=walloc(64,64),  *t0P3=walloc(64,32);
    __bf16 *c0P1=walloc(96,64),  *c0P2=walloc(64,64),  *c0P3=walloc(64,64);
    __bf16 *s1P1=walloc(160,64), *s1P2=walloc(64,64),  *s1P3=walloc(64,32);
    __bf16 *t1P1=walloc(224,64), *t1P2=walloc(64,64),  *t1P3=walloc(64,32);
    __bf16 *c1P1=walloc(128,64), *c1P2=walloc(64,64),  *c1P3=walloc(64,64);
    __bf16 *tgP1=walloc(96,128), *tgP2=walloc(128,128),*tgP3=walloc(128,64);

    auto pack = [&](const float* W, __bf16* dst, int dA, int dApad, int dB,
                    int dout, int KDIM) {
        long total = (long)(KDIM / 32) * (dout / 16) * 512;
        int blocks = (int)((total + 255) / 256);
        k_pack_w<<<blocks, 256, 0, stream>>>(W, dst, dA, dApad, dB, dout, KDIM);
    };
    // layer-1 packs carry the concat/pad row remap; layers 2/3 are dense.
    pack(s0W1, s0P1, 35, 48, 35, 64, 96);
    pack(s0W2, s0P2, 64, 64, 0, 64, 64);
    pack(s0W3, s0P3, 64, 64, 0, 32, 64);
    pack(t0W1, t0P1, 72, 80, 72, 64, 160);
    pack(t0W2, t0P2, 64, 64, 0, 64, 64);
    pack(t0W3, t0P3, 64, 64, 0, 32, 64);
    pack(c0W1, c0P1, 96, 96, 0, 64, 96);
    pack(c0W2, c0P2, 64, 64, 0, 64, 64);
    pack(c0W3, c0P3, 64, 64, 0, 64, 64);
    pack(s1W1, s1P1, 67, 80, 67, 64, 160);
    pack(s1W2, s1P2, 64, 64, 0, 64, 64);
    pack(s1W3, s1P3, 64, 64, 0, 32, 64);
    pack(t1W1, t1P1, 104, 112, 104, 64, 224);
    pack(t1W2, t1P2, 64, 64, 0, 64, 64);
    pack(t1W3, t1P3, 64, 64, 0, 32, 64);
    pack(c1W1, c1P1, 128, 128, 0, 64, 128);
    pack(c1W2, c1P2, 64, 64, 0, 64, 64);
    pack(c1W3, c1P3, 64, 64, 0, 64, 64);
    pack(tgW1, tgP1, 8, 16, 72, 128, 96);
    pack(tgW2, tgP2, 128, 128, 0, 128, 128);
    pack(tgW3, tgP3, 128, 128, 0, 64, 128);

    k_enc_time<<<(BN + 255) / 256, 256, 0, stream>>>(time_pts, et, BN);
    k_enc_query<<<(BM + 255) / 256, 256, 0, stream>>>(query_pts, qenc, BM);

    auto cat = [&](__bf16* dst, int DP, long R,
                   const float* a, int da, const float* b, int db,
                   const float* c, int dc) {
        long total = R * DP;
        int blocks = (int)((total + 255) / 256);
        if (blocks > 16384) blocks = 16384;
        k_concat3<<<blocks, 256, 0, stream>>>(dst, DP, R, a, da, b, db, c, dc);
    };

    // ---------- layer 0 ----------
    cat(nodeA, 48, BN, space_pts, 3, data, 32, nullptr, 0);
    edge_mlp<16, 48, 48, 64, 64, 32, Nn, Nn>
        <<<(BN * 16) / 128, 256, smem_bytes(96, 64, 64, 32), stream>>>(
            nodeA, nodeA, space_idx, space_valid,
            s0P1, s0b1, s0P2, s0b2, s0P3, s0b3, sbuf);

    cat(nodeA, 80, BN, et, 8, data, 32, sbuf, 32);
    edge_mlp<8, 80, 80, 64, 64, 32, Nn, Nn>
        <<<(BN * 8) / 128, 256, smem_bytes(160, 64, 64, 32), stream>>>(
            nodeA, nodeA, time_idx, time_valid,
            t0P1, t0b1, t0P2, t0b2, t0P3, t0b3, tbuf);

    cat(nodeA, 96, BN, data, 32, sbuf, 32, tbuf, 32);
    edge_mlp<1, 96, 0, 64, 64, 64, Nn, Nn>
        <<<BN / 128, 256, smem_bytes(96, 64, 64, 64), stream>>>(
            nodeA, nullptr, nullptr, nullptr,
            c0P1, c0b1, c0P2, c0b2, c0P3, c0b3, key);

    // ---------- layer 1 ----------
    cat(nodeA, 80, BN, space_pts, 3, key, 64, nullptr, 0);
    edge_mlp<16, 80, 80, 64, 64, 32, Nn, Nn>
        <<<(BN * 16) / 128, 256, smem_bytes(160, 64, 64, 32), stream>>>(
            nodeA, nodeA, space_idx, space_valid,
            s1P1, s1b1, s1P2, s1b2, s1P3, s1b3, sbuf);

    cat(nodeA, 112, BN, et, 8, key, 64, sbuf, 32);
    edge_mlp<8, 112, 112, 64, 64, 32, Nn, Nn>
        <<<(BN * 8) / 128, 256, smem_bytes(224, 64, 64, 32), stream>>>(
            nodeA, nodeA, time_idx, time_valid,
            t1P1, t1b1, t1P2, t1b2, t1P3, t1b3, tbuf);

    cat(nodeA, 128, BN, key, 64, sbuf, 32, tbuf, 32);
    edge_mlp<1, 128, 0, 64, 64, 64, Nn, Nn>
        <<<BN / 128, 256, smem_bytes(128, 64, 64, 64), stream>>>(
            nodeA, nullptr, nullptr, nullptr,
            c1P1, c1b1, c1P2, c1b2, c1P3, c1b3, key);

    // ---------- target head ----------
    cat(nodeA, 80, BN, et, 8, key, 64, nullptr, 0);
    edge_mlp<8, 16, 80, 128, 128, 64, Mm, Nn>
        <<<(BM * 8) / 128, 256, smem_bytes(96, 128, 128, 64), stream>>>(
            qenc, nodeA, target_idx, target_valid,
            tgP1, tgb1, tgP2, tgb2, tgP3, tgb3, (float*)d_out);
}
